// RNN_8392366096807
// MI455X (gfx1250) — matrix operand order repaired and checked by reading, not compiled, not run on hardware
//
#include <hip/hip_runtime.h>

typedef __attribute__((ext_vector_type(16))) _Float16 v16h;
typedef __attribute__((ext_vector_type(8)))  _Float16 v8h;
typedef __attribute__((ext_vector_type(2)))  _Float16 h2v;
typedef __attribute__((ext_vector_type(8)))  float    v8f;

#define SEQ   256
#define DEMB  64
#define DHID  256
#define KDIM  320           // DEMB + DHID
#define KT    10            // 320 / 32 K-tiles
#define ROWS  16            // batch rows per workgroup
#define ASTR  328           // padded LDS stride (f16 elems) to avoid bank conflicts

// fast sigmoid: v_exp_f32 + v_rcp_f32
__device__ __forceinline__ float fast_sigmoid(float z) {
  return __builtin_amdgcn_rcpf(1.f + __expf(-z));
}

// ---------------------------------------------------------------------------
// Prep: pack Wh (320x256 f32, row-major) into f16 WMMA B-fragment layout.
//   lane <  16 : K = kt*32 + e,      N = nt*16 + lane
//   lane >= 16 : K = kt*32 + 16 + e, N = nt*16 + lane - 16
// ---------------------------------------------------------------------------
__global__ __launch_bounds__(512)
void pack_wh_f16(const float* __restrict__ Wh, _Float16* __restrict__ wpk) {
  int pid = blockIdx.x * 512 + threadIdx.x;
  if (pid >= KDIM * DHID) return;
  int blk    = pid >> 9;
  int within = pid & 511;
  int lane   = within >> 4;
  int e      = within & 15;
  int nt = blk / KT;
  int kt = blk % KT;
  int K, N;
  if (lane < 16) { K = kt * 32 + e;      N = nt * 16 + lane;      }
  else           { K = kt * 32 + 16 + e; N = nt * 16 + lane - 16; }
  wpk[pid] = (_Float16)Wh[K * DHID + N];
}

// ---------------------------------------------------------------------------
// Main: whole 256-step scan per workgroup. 16 waves; wave w owns N-tile w.
// Software-pipelined: x_{t+1} gather + idx_{t+2} load overlap step t's WMMAs.
// ---------------------------------------------------------------------------
__global__ __launch_bounds__(512)
void rnn_scan(const int* __restrict__ data, const float* __restrict__ emb,
              const _Float16* __restrict__ wpk, const float* __restrict__ bh,
              const float* __restrict__ Wo, const float* __restrict__ bo,
              float* __restrict__ out) {
  __shared__ __align__(16) _Float16 Ash[ROWS][ASTR];     // [x(0:64) | h(64:320)] f16
  __shared__ __align__(8)  unsigned char maskb[2][ROWS]; // double-buffered row masks

  const int tid  = threadIdx.x;
  const int lane = tid & 31;
  const int w    = tid >> 5;            // wave id: phase-1 row AND gemm N-tile
  const int row0 = blockIdx.x * ROWS;
  const int half = lane >> 4;
  const int m    = lane & 15;

  // --- preload this wave's B fragments (Wh, packed f16) into registers ---
  v16h bfrag[KT];
  #pragma unroll
  for (int kt = 0; kt < KT; ++kt) {
    const v16h* bp = (const v16h*)(wpk + (size_t)(w * KT + kt) * 512);
    bfrag[kt] = bp[lane];
  }
  const float bhv = bh[w * 16 + m];

  const int rowbase = (row0 + w) * SEQ;

  // --- 'last' embedding: 2 elements per lane, register-resident ---
  int il = data[rowbase + (SEQ - 1)];
  float2 lv = *(const float2*)(emb + (size_t)il * DEMB + 2 * lane);

  // --- zero h region of Ash + register h fragment ---
  #pragma unroll
  for (int j = 0; j < 8; ++j)
    Ash[w][DEMB + lane + j * 32] = (_Float16)0.f;
  float hreg[8] = {0.f, 0.f, 0.f, 0.f, 0.f, 0.f, 0.f, 0.f};

  // --- pipeline prologue: x for t=0, idx for t=1 ---
  int idx_c = data[rowbase + 0];
  int idx_n = data[rowbase + 1];
  float2 xc = *(const float2*)(emb + (size_t)idx_c * DEMB + 2 * lane);

  __syncthreads();

  for (int t = 0; t < SEQ; ++t) {
    // ---- phase 1: exact f32 mask (ballot) + stage x as packed f16 ----
    bool ne = (xc.x != lv.x) | (xc.y != lv.y);
    h2v xp; xp.x = (_Float16)xc.x; xp.y = (_Float16)xc.y;
    *(h2v*)(&Ash[w][2 * lane]) = xp;                    // one ds_store_b32
#if __has_builtin(__builtin_amdgcn_ballot_w32)
    unsigned bal = __builtin_amdgcn_ballot_w32(ne);
#else
    unsigned bal = ne ? 1u : 0u;
    bal |= __shfl_xor(bal, 16); bal |= __shfl_xor(bal, 8);
    bal |= __shfl_xor(bal, 4);  bal |= __shfl_xor(bal, 2);  bal |= __shfl_xor(bal, 1);
#endif
    if (lane == 0) maskb[t & 1][w] = (unsigned char)(bal != 0u);
    __syncthreads();

    // ---- phase 2a: mask bytes (ds_load_b64) + ALL A fragments ----
    uint2 mw = *(const uint2*)(&maskb[t & 1][half * 8]);
    v16h afrag[KT];
    #pragma unroll
    for (int kt = 0; kt < KT; ++kt) {
      const _Float16* p = &Ash[m][kt * 32 + half * 16];  // elems 0-15: K=16h+e
      v8h lo = *(const v8h*)p;
      v8h hi = *(const v8h*)(p + 8);
      afrag[kt] = __builtin_shufflevector(lo, hi,
          0, 1, 2, 3, 4, 5, 6, 7, 8, 9, 10, 11, 12, 13, 14, 15);
    }
    __syncthreads();   // A reads done -> safe to overwrite Ash in epilogue

    // ---- prefetch (overlaps with WMMAs): x_{t+1} gather, idx_{t+2} ----
    int t2 = (t + 2 < SEQ) ? (t + 2) : (SEQ - 1);
    float2 xn = *(const float2*)(emb + (size_t)idx_n * DEMB + 2 * lane);
    int idx_n2 = data[rowbase + t2];

    // ---- phase 2b: 10x v_wmma_f32_16x16x32_f16, K = 320 ----
    v8f acc = {};
    #pragma unroll
    for (int kt = 0; kt < KT; ++kt)
      acc = __builtin_amdgcn_wmma_f32_16x16x32_f16(
          false, afrag[kt], false, bfrag[kt], (short)0, acc, false, false);

    // ---- phase 3: bias + sigmoid + branchless masked h update ----
    #pragma unroll
    for (int e = 0; e < 8; ++e) {
      int row = e + half * 8;                 // M of acc element e for this half-wave
      float s = fast_sigmoid(acc[e] + bhv);
      unsigned bit = (e < 4) ? (mw.x >> (8 * e)) : (mw.y >> (8 * (e - 4)));
      float h = (bit & 1u) ? s : hreg[e];     // v_cndmask, no branch
      hreg[e] = h;
      Ash[row][DEMB + w * 16 + m] = (_Float16)h;
    }

    xc = xn;
    idx_n = idx_n2;
  }
  __syncthreads();

  // ---- output projection: wave w reduces row w: sigmoid(h . Wo + bo) ----
  float partial = 0.f;
  #pragma unroll
  for (int j = 0; j < 8; ++j) {
    int k = lane + j * 32;
    partial += (float)Ash[w][DEMB + k] * Wo[k];
  }
  partial += __shfl_xor(partial, 16); partial += __shfl_xor(partial, 8);
  partial += __shfl_xor(partial, 4);  partial += __shfl_xor(partial, 2);
  partial += __shfl_xor(partial, 1);
  if (lane == 0)
    out[row0 + w] = fast_sigmoid(partial + bo[0]);
}

// ---------------------------------------------------------------------------
extern "C" void kernel_launch(void* const* d_in, const int* in_sizes, int n_in,
                              void* d_out, int out_size, void* d_ws, size_t ws_size,
                              hipStream_t stream) {
  const int*   data = (const int*)d_in[0];
  const float* emb  = (const float*)d_in[1];
  const float* Wh   = (const float*)d_in[2];
  const float* bh   = (const float*)d_in[3];
  const float* Wo   = (const float*)d_in[4];
  const float* bo   = (const float*)d_in[5];
  float*       out  = (float*)d_out;
  _Float16*    wpk  = (_Float16*)d_ws;   // 320*256 f16 = 160 KB of scratch

  int batch = in_sizes[0] / SEQ;

  pack_wh_f16<<<(KDIM * DHID + 511) / 512, 512, 0, stream>>>(Wh, wpk);
  rnn_scan<<<batch / ROWS, 512, 0, stream>>>(data, emb, wpk, bh, Wo, bo, out);

  (void)n_in; (void)out_size; (void)ws_size;
}
